// CrossAttentionBlock_12257836662867
// MI455X (gfx1250) — matrix operand order repaired and checked by reading, compiled, not run on hardware
//
#include <hip/hip_runtime.h>

typedef __attribute__((ext_vector_type(16))) __bf16 v16bf;
typedef __attribute__((ext_vector_type(8)))  __bf16 v8bf;
typedef __attribute__((ext_vector_type(8)))  float  v8f;
typedef __attribute__((ext_vector_type(4)))  float  v4f;
typedef __attribute__((ext_vector_type(4)))  int    v4i;

#define BATCH 8
#define LQ 1024
#define LK 2048
#define DM 512

#if defined(__has_builtin)
#if __has_builtin(__builtin_amdgcn_global_load_async_to_lds_b128)
#define HAVE_ASYNC_LDS 1
#endif
#endif
#ifndef HAVE_ASYNC_LDS
#define HAVE_ASYNC_LDS 0
#endif

// ---- bf16 <-> f32 (native v_cvt_pk_bf16_f32 path) ----
__device__ __forceinline__ __bf16 f2bf(float f) { return (__bf16)f; }
__device__ __forceinline__ float bf2f(__bf16 h) { return (float)h; }

struct bfpair { v8bf lo, hi; };
__device__ __forceinline__ v16bf make_frag(v8bf lo, v8bf hi) {
  bfpair p{lo, hi};
  return __builtin_bit_cast(v16bf, p);
}
__device__ __forceinline__ v8bf cvt8(v4f a, v4f b) {
  v8bf r;
  r[0] = f2bf(a[0]); r[1] = f2bf(a[1]); r[2] = f2bf(a[2]); r[3] = f2bf(a[3]);
  r[4] = f2bf(b[0]); r[5] = f2bf(b[1]); r[6] = f2bf(b[2]); r[7] = f2bf(b[3]);
  return r;
}

// ---- WMMA fragment loaders (wave32; contiguous K order chosen here, A and B alike) ----
// A (16x32): lane row m = lane%16; elems 0..15 -> K=b0..b0+15,
// b0 = 16*(lane>=16).   B (32x16): lane col n = lane%16; 16 contiguous K at 16*(lane>=16).
__device__ __forceinline__ v16bf load_a_f32(const float* row, int b0) {
  v4f x0 = *(const v4f*)(row + b0);
  v4f x1 = *(const v4f*)(row + b0 + 4);
  v4f x2 = *(const v4f*)(row + b0 + 8);
  v4f x3 = *(const v4f*)(row + b0 + 12);
  return make_frag(cvt8(x0, x1), cvt8(x2, x3));
}
__device__ __forceinline__ v16bf load_b_f32(const float* p) {
  v4f x0 = *(const v4f*)p;
  v4f x1 = *(const v4f*)(p + 4);
  v4f x2 = *(const v4f*)(p + 8);
  v4f x3 = *(const v4f*)(p + 12);
  return make_frag(cvt8(x0, x1), cvt8(x2, x3));
}
__device__ __forceinline__ v16bf load_a_any(const __bf16* row, int b0) {
  v8bf lo = *(const v8bf*)(row + b0);
  v8bf hi = *(const v8bf*)(row + b0 + 8);
  return make_frag(lo, hi);
}
__device__ __forceinline__ v16bf load_b_bf(const __bf16* p) {
  return *(const v16bf*)p;  // 32B contiguous
}

#define WMMA_BF16(A, B, C) \
  __builtin_amdgcn_wmma_f32_16x16x32_bf16(false, (A), false, (B), (short)0, (C), false, false)

// ---- Stage a 16 x 512 bf16 A-tile (16KB) into LDS, shared by all 8 waves ----
__device__ __forceinline__ void stage_a(const __bf16* g0, size_t stride, __bf16* lds) {
  __syncthreads();                       // previous readers done before overwrite
  const int t = threadIdx.x;
  // 16 rows x 512 elems = 1024 x 16B chunks; 4 per thread
#pragma unroll
  for (int i = 0; i < 4; ++i) {
    const int c   = t + i * 256;
    const int row = c >> 6;              // 64 chunks per row
    const int col = (c & 63) << 3;
    const __bf16* g = g0 + (size_t)row * stride + col;
    __bf16* l = lds + row * 512 + col;
#if HAVE_ASYNC_LDS
    __builtin_amdgcn_global_load_async_to_lds_b128(
        (v4i*)const_cast<__bf16*>(g), (v4i*)l, 0, 0);
#else
    *(v8bf*)l = *(const v8bf*)g;
#endif
  }
#if HAVE_ASYNC_LDS
  asm volatile("s_wait_asynccnt 0" ::: "memory");
#endif
  __syncthreads();
}

// ================= Kernel 1: projection GEMM  Out = (X * W^T + bias) * alpha =================
// Block = (b, mtile); each of 8 waves owns a 16x64 register tile; block covers N=512.
// Software-pipelined: next iteration's loads/cvts issue before current WMMAs so
// v_cvt never writes an in-flight WMMA's source registers (kills the 4-NOP WAR).
__global__ void proj_gemm(const float* __restrict__ X, const float* __restrict__ W,
                          const float* __restrict__ bias, __bf16* __restrict__ Out,
                          int L, float alpha, int transposed) {
  const int wave = threadIdx.x >> 5;
  const int lane = threadIdx.x & 31;
  const int l16  = lane & 15;
  const int half = lane >> 4;
  const int mtiles = L >> 4;
  const int b     = blockIdx.x / mtiles;
  const int mtile = blockIdx.x % mtiles;
  const int mbase = mtile << 4;
  const int nbase0 = wave << 6;
  const int b0 = half * 16;

  const float* arow = X + (size_t)(b * L + mbase + l16) * DM;
  const float* brow[4];
#pragma unroll
  for (int j = 0; j < 4; ++j)
    brow[j] = W + (size_t)(nbase0 + j * 16 + l16) * DM + half * 16;

  v8f acc[4] = {};
  v16bf A = load_a_f32(arow, b0);
  v16bf Bf[4];
#pragma unroll
  for (int j = 0; j < 4; ++j) Bf[j] = load_b_f32(brow[j]);

#pragma unroll
  for (int kb = 32; kb <= DM; kb += 32) {
    v16bf An = A;
    v16bf Bn[4];
    if (kb < DM) {
      __builtin_prefetch(arow + kb + 96, 0, 1);
      An = load_a_f32(arow + kb, b0);
#pragma unroll
      for (int j = 0; j < 4; ++j) Bn[j] = load_b_f32(brow[j] + kb);
    }
#pragma unroll
    for (int j = 0; j < 4; ++j) acc[j] = WMMA_BF16(A, Bf[j], acc[j]);
    if (kb < DM) {
      A = An;
#pragma unroll
      for (int j = 0; j < 4; ++j) Bf[j] = Bn[j];
    }
  }
#pragma unroll
  for (int j = 0; j < 4; ++j) {
    const int nb = nbase0 + j * 16;
    const float bb = bias[nb + l16];
    if (!transposed) {
      size_t base = (size_t)(b * L + mbase + half * 8) * DM + nb + l16;
#pragma unroll
      for (int r = 0; r < 8; ++r)
        Out[base + (size_t)r * DM] = f2bf((acc[j][r] + bb) * alpha);
    } else {
      v8bf o;
#pragma unroll
      for (int r = 0; r < 8; ++r) o[r] = f2bf((acc[j][r] + bb) * alpha);
      *(v8bf*)(Out + (size_t)(b * DM + nb + l16) * L + mbase + half * 8) = o;
    }
  }
}

// ================= Kernel 2: S = Q * K^T  (bf16 out, 1/sqrt(D) folded into Q) =================
// A-tile (16x512 = whole K-dim) staged in LDS once, shared by 8 waves; pipelined loop.
__global__ void scores_gemm(const __bf16* __restrict__ Qb, const __bf16* __restrict__ Kb,
                            __bf16* __restrict__ Sb) {
  __shared__ __bf16 As[16 * 512];
  const int wave = threadIdx.x >> 5;
  const int lane = threadIdx.x & 31;
  const int l16  = lane & 15;
  const int half = lane >> 4;
  const int mtiles = LQ >> 4;
  const int b     = blockIdx.x / mtiles;
  const int mtile = blockIdx.x % mtiles;
  const int mbase = mtile << 4;
  const int nbase0 = (blockIdx.y << 9) + (wave << 6);
  const int b0 = half * 16;

  stage_a(Qb + (size_t)(b * LQ + mbase) * DM, DM, As);

  const __bf16* alds = As + l16 * 512;
  const __bf16* brow[4];
#pragma unroll
  for (int j = 0; j < 4; ++j)
    brow[j] = Kb + (size_t)(b * LK + nbase0 + j * 16 + l16) * DM + half * 16;

  v8f acc[4] = {};
  v16bf A = load_a_any(alds, b0);          // ds_load
  v16bf Bf[4];
#pragma unroll
  for (int j = 0; j < 4; ++j) Bf[j] = load_b_bf(brow[j]);

#pragma unroll
  for (int kb = 32; kb <= DM; kb += 32) {
    v16bf An = A;
    v16bf Bn[4];
    if (kb < DM) {
      An = load_a_any(alds + kb, b0);
#pragma unroll
      for (int j = 0; j < 4; ++j) Bn[j] = load_b_bf(brow[j] + kb);
    }
#pragma unroll
    for (int j = 0; j < 4; ++j) acc[j] = WMMA_BF16(A, Bf[j], acc[j]);
    if (kb < DM) {
      A = An;
#pragma unroll
      for (int j = 0; j < 4; ++j) Bf[j] = Bn[j];
    }
  }
#pragma unroll
  for (int j = 0; j < 4; ++j) {
    size_t base = (size_t)(b * LQ + mbase + half * 8) * LK + nbase0 + j * 16 + l16;
#pragma unroll
    for (int r = 0; r < 8; ++r)
      Sb[base + (size_t)r * LK] = f2bf(acc[j][r]);
  }
}

// ================= Kernel 3: in-place row softmax over Lk=2048 =================
__global__ void softmax_inplace(__bf16* __restrict__ S) {
  __shared__ float red[8];
  __bf16* row = S + (size_t)blockIdx.x * LK;
  const int t = threadIdx.x;

  v8bf chunk = *(const v8bf*)(row + t * 8);
  float v[8];
  float m = -3.0e38f;
#pragma unroll
  for (int i = 0; i < 8; ++i) { v[i] = bf2f(chunk[i]); m = fmaxf(m, v[i]); }
#pragma unroll
  for (int off = 16; off > 0; off >>= 1) m = fmaxf(m, __shfl_xor(m, off, 32));
  if ((t & 31) == 0) red[t >> 5] = m;
  __syncthreads();
  m = red[0];
#pragma unroll
  for (int i = 1; i < 8; ++i) m = fmaxf(m, red[i]);
  __syncthreads();

  float s = 0.f;
#pragma unroll
  for (int i = 0; i < 8; ++i) { v[i] = __expf(v[i] - m); s += v[i]; }
#pragma unroll
  for (int off = 16; off > 0; off >>= 1) s += __shfl_xor(s, off, 32);
  if ((t & 31) == 0) red[t >> 5] = s;
  __syncthreads();
  s = 0.f;
#pragma unroll
  for (int i = 0; i < 8; ++i) s += red[i];

  const float inv = 1.0f / s;
  v8bf o;
#pragma unroll
  for (int i = 0; i < 8; ++i) o[i] = f2bf(v[i] * inv);
  *(v8bf*)(row + t * 8) = o;
}

// ================= Kernel 4: Out = P * V  (V stored transposed [B, D, Lk]) =================
// A rows are 2048 wide: staged in LDS in 4 chunks of 16x512; pipelined inner loop.
__global__ void pv_gemm(const __bf16* __restrict__ P, const __bf16* __restrict__ Vt,
                        float* __restrict__ Out) {
  __shared__ __bf16 As[16 * 512];
  const int wave = threadIdx.x >> 5;
  const int lane = threadIdx.x & 31;
  const int l16  = lane & 15;
  const int half = lane >> 4;
  const int mtiles = LQ >> 4;
  const int b     = blockIdx.x / mtiles;
  const int mtile = blockIdx.x % mtiles;
  const int mbase = mtile << 4;
  const int nbase0 = wave << 6;
  const int b0 = half * 16;

  const __bf16* agrow = P + (size_t)(b * LQ + mbase) * LK;
  const __bf16* alds  = As + l16 * 512;
  const __bf16* brow[4];
#pragma unroll
  for (int j = 0; j < 4; ++j)
    brow[j] = Vt + (size_t)(b * DM + nbase0 + j * 16 + l16) * LK + half * 16;

  v8f acc[4] = {};
  for (int c = 0; c < LK; c += 512) {
    stage_a(agrow + c, LK, As);
    v16bf A = load_a_any(alds, b0);        // ds_load
    v16bf Bf[4];
#pragma unroll
    for (int j = 0; j < 4; ++j) Bf[j] = load_b_bf(brow[j] + c);

#pragma unroll
    for (int kb = 32; kb <= 512; kb += 32) {
      v16bf An = A;
      v16bf Bn[4];
      if (kb < 512) {
        An = load_a_any(alds + kb, b0);
#pragma unroll
        for (int j = 0; j < 4; ++j) Bn[j] = load_b_bf(brow[j] + c + kb);
      }
#pragma unroll
      for (int j = 0; j < 4; ++j) acc[j] = WMMA_BF16(A, Bf[j], acc[j]);
      if (kb < 512) {
        A = An;
#pragma unroll
        for (int j = 0; j < 4; ++j) Bf[j] = Bn[j];
      }
    }
  }
#pragma unroll
  for (int j = 0; j < 4; ++j) {
    size_t base = (size_t)(b * LQ + mbase + half * 8) * DM + nbase0 + j * 16 + l16;
#pragma unroll
    for (int r = 0; r < 8; ++r)
      Out[base + (size_t)r * DM] = acc[j][r];
  }
}

extern "C" void kernel_launch(void* const* d_in, const int* in_sizes, int n_in,
                              void* d_out, int out_size, void* d_ws, size_t ws_size,
                              hipStream_t stream) {
  (void)in_sizes; (void)n_in; (void)out_size; (void)ws_size;
  const float* query = (const float*)d_in[0];
  const float* key   = (const float*)d_in[1];
  const float* Wq    = (const float*)d_in[2];
  const float* bq    = (const float*)d_in[3];
  const float* Wk    = (const float*)d_in[4];
  const float* bk    = (const float*)d_in[5];
  const float* Wv    = (const float*)d_in[6];
  const float* bv    = (const float*)d_in[7];
  float* out = (float*)d_out;

  __bf16* Qb  = (__bf16*)d_ws;                          // [B, LQ, DM]
  __bf16* Kb  = Qb  + (size_t)BATCH * LQ * DM;          // [B, LK, DM]
  __bf16* Vtb = Kb  + (size_t)BATCH * LK * DM;          // [B, DM, LK]
  __bf16* Sb  = Vtb + (size_t)BATCH * DM * LK;          // [B, LQ, LK] (scores -> probs, in place)

  const float inv_sqrt_d = 0.04419417382415922f;        // 1/sqrt(512), folded into Q
  dim3 blk(256);

  proj_gemm<<<dim3(BATCH * (LQ / 16)), blk, 0, stream>>>(query, Wq, bq, Qb,  LQ, inv_sqrt_d, 0);
  proj_gemm<<<dim3(BATCH * (LK / 16)), blk, 0, stream>>>(key,   Wk, bk, Kb,  LK, 1.0f,       0);
  proj_gemm<<<dim3(BATCH * (LK / 16)), blk, 0, stream>>>(key,   Wv, bv, Vtb, LK, 1.0f,       1);
  scores_gemm<<<dim3(BATCH * (LQ / 16), 4), blk, 0, stream>>>(Qb, Kb, Sb);
  softmax_inplace<<<dim3(BATCH * LQ), blk, 0, stream>>>(Sb);
  pv_gemm<<<dim3(BATCH * (LQ / 16)), blk, 0, stream>>>(Sb, Vtb, out);
}
